// SparseAttention_56100862820421
// MI455X (gfx1250) — matrix operand order repaired and checked by reading, not compiled, not run on hardware
//
#include <hip/hip_runtime.h>
#include <stdint.h>

// ---------------------------------------------------------------------------
// SparseAttention for MI455X (gfx1250): wave32, V_WMMA_F32_16X16X32_F16,
// async global->LDS staging (GLOBAL_LOAD_ASYNC_TO_LDS_B128 + s_wait_asynccnt).
//
// Only a 64x64 attention block per (b,h) is used; heavy work is
// v-proj (16384x768x768), per-head 64-wide attn apply, out-proj + residual.
// Weights are pre-transposed into panel-contiguous f16 ([head][kchunk][n][k])
// once per launch so all hot-loop staging is pure 16B/lane async block copy.
// ---------------------------------------------------------------------------

typedef _Float16 h16v __attribute__((ext_vector_type(16)));
typedef _Float16 h8v  __attribute__((ext_vector_type(8)));
typedef float    f8v  __attribute__((ext_vector_type(8)));

#define NH   12
#define DIM  768
#define SEQ  2048
#define NB   8
#define QK_SCALE 0.125f    // 1/sqrt(64)

#define KS64 72            // padded k-stride (halves) for 64-wide tiles
#define KSY  776           // padded k-stride (halves) for 768-wide tiles
#define PANEL 4096         // one 64x64 f16 panel, in halves
#define MATP (144 * PANEL) // 12x12 panels per 768x768 matrix (589824 halves)

// ---- async global->LDS copy (CDNA5): 16B per lane, ASYNCcnt tracked -------
__device__ __forceinline__ void async_ld_b128(_Float16* lds_dst,
                                              const _Float16* src) {
  // Generic pointers to __shared__ carry the LDS offset in the low 32 bits.
  uint32_t off = (uint32_t)(uintptr_t)lds_dst;
  asm volatile("global_load_async_to_lds_b128 %0, %1, off"
               :: "v"(off), "v"(src)
               : "memory");
}
__device__ __forceinline__ void wait_async() {
  asm volatile("s_wait_asynccnt 0x0" ::: "memory");
}

// Stage one 64x64 f16 panel (panel-contiguous in global) into padded LDS
// tile [n][KS64]. 512 chunks of 16B; all lanes active (EXEC all ones).
__device__ __forceinline__ void stage_panel(_Float16* w_lds,
                                            const _Float16* pan, int tid,
                                            int nthreads) {
#pragma unroll
  for (int e = tid; e < 512; e += nthreads) {
    int n = e >> 3, s = e & 7;
    async_ld_b128(w_lds + n * KS64 + s * 8, pan + e * 8);
  }
  wait_async();
}

// ---- WMMA f16 fragment load from LDS tile stored [row][k] -----------------
// ISA 16-bit A (16x32): lanes 0-15 hold K{0..7,16..23}, lanes 16-31 hold
// K{8..15,24..31} -> two ds_load_b128. B (32x16) mirrors with N for M,
// so a [n][k] tile serves B fragments.
__device__ __forceinline__ h16v frag_ld(const _Float16* p) {
  h8v lo = *(const h8v*)(p);
  h8v hi = *(const h8v*)(p + 16);
  h16v r;
#pragma unroll
  for (int i = 0; i < 8; ++i) { r[i] = lo[i]; r[8 + i] = hi[i]; }
  return r;
}

__device__ __forceinline__ f8v wmma_f16(h16v a, h16v b, f8v c) {
  return __builtin_amdgcn_wmma_f32_16x16x32_f16(
      false, a, false, b, (short)0, c, false, false);
}

// ---------------------------------------------------------------------------
// Kernel 0: pre-transpose weights to panel-contiguous f16.
//   dst[((j*12+ks)*64+n)*64+k] = (f16) W[ks*64+k][coloff + j*64 + n]
// ---------------------------------------------------------------------------
__global__ __launch_bounds__(256) void prep_weights(
    const float* __restrict__ Wq, const float* __restrict__ Wkv,
    const float* __restrict__ Wo, _Float16* __restrict__ WqT,
    _Float16* __restrict__ WkT, _Float16* __restrict__ WvT,
    _Float16* __restrict__ WoT) {
  int idx = blockIdx.x * 256 + threadIdx.x;
  if (idx >= 4 * MATP) return;
  int m = idx / MATP;
  int r = idx - m * MATP;
  int k = r & 63;
  int n = (r >> 6) & 63;
  int p = r >> 12;           // 0..143
  int ks = p % 12, j = p / 12;

  const float* src; _Float16* dst; int ld, co;
  if (m == 0)      { src = Wq;  dst = WqT; ld = DIM;     co = 0;   }
  else if (m == 1) { src = Wkv; dst = WkT; ld = 2 * DIM; co = 0;   }
  else if (m == 2) { src = Wkv; dst = WvT; ld = 2 * DIM; co = DIM; }
  else             { src = Wo;  dst = WoT; ld = DIM;     co = 0;   }
  dst[r] = (_Float16)src[(size_t)(ks * 64 + k) * ld + co + j * 64 + n];
}

// ---- 64x64 projection: out_lds = src[0:64,:768] @ W_panels(head) + bias ---
__device__ __forceinline__ void project64(
    const float* __restrict__ src, const _Float16* __restrict__ WT,
    const float* __restrict__ bias, int bias0, float scale,
    _Float16* out_lds, _Float16* a_lds, _Float16* w_lds, int tid) {
  const int lane = tid & 31, wave = tid >> 5;
  const int mt = wave, col0 = lane & 15, grp8 = (lane >> 4) * 8;
  const int rb = (lane >> 4) * 8;
  f8v zero = {};
  f8v acc[4] = {zero, zero, zero, zero};

  for (int ks = 0; ks < DIM / 64; ++ks) {
    __syncthreads();
    // stage A chunk f32->f16
#pragma unroll 4
    for (int e = tid; e < 64 * 64; e += 128) {
      int r = e >> 6, c = e & 63;
      a_lds[r * KS64 + c] = (_Float16)src[(size_t)r * DIM + ks * 64 + c];
    }
    stage_panel(w_lds, WT + ks * PANEL, tid, 128);
    __syncthreads();
    const _Float16* ap = a_lds + (mt * 16 + col0) * KS64 + grp8;
    h16v a0 = frag_ld(ap), a1 = frag_ld(ap + 32);
#pragma unroll
    for (int nt = 0; nt < 4; ++nt) {
      const _Float16* wp = w_lds + (nt * 16 + col0) * KS64 + grp8;
      h16v w0 = frag_ld(wp), w1 = frag_ld(wp + 32);
      acc[nt] = wmma_f16(a0, w0, acc[nt]);
      acc[nt] = wmma_f16(a1, w1, acc[nt]);
    }
  }
  __syncthreads();
#pragma unroll
  for (int nt = 0; nt < 4; ++nt) {
    float bv = bias[bias0 + nt * 16 + col0];
#pragma unroll
    for (int i = 0; i < 8; ++i)
      out_lds[(mt * 16 + rb + i) * KS64 + nt * 16 + col0] =
          (_Float16)((acc[nt][i] + bv) * scale);
  }
  __syncthreads();
}

// ---------------------------------------------------------------------------
// Kernel 1: per (b,h) attention probabilities, stored transposed as f16.
// PT[z][s] = P[s][z] is exactly the B-fragment [n][k] layout for V@P.
// ---------------------------------------------------------------------------
__global__ __launch_bounds__(128) void attn_prob_kernel(
    const float* __restrict__ x, const float* __restrict__ y,
    const _Float16* __restrict__ WqT, const float* __restrict__ bq,
    const _Float16* __restrict__ WkT, const float* __restrict__ bkv,
    _Float16* __restrict__ attnT) {
  extern __shared__ char smem[];
  _Float16* q_lds = (_Float16*)smem;        // [64][72]
  _Float16* k_lds = q_lds + 64 * KS64;      // [64][72]
  _Float16* a_lds = k_lds + 64 * KS64;      // staging
  _Float16* w_lds = a_lds + 64 * KS64;      // staging

  const int b = blockIdx.x / NH, h = blockIdx.x % NH;
  const int tid = threadIdx.x, lane = tid & 31, wave = tid >> 5;
  const int mt = wave, col0 = lane & 15, grp8 = (lane >> 4) * 8;
  const int rb = (lane >> 4) * 8;

  const float* xs = x + (size_t)b * SEQ * DIM;
  const float* ys = y + (size_t)b * SEQ * DIM;

  project64(xs, WqT + (size_t)h * 12 * PANEL, bq, h * 64, QK_SCALE,
            q_lds, a_lds, w_lds, tid);
  project64(ys, WkT + (size_t)h * 12 * PANEL, bkv, h * 64, 1.0f,
            k_lds, a_lds, w_lds, tid);

  // scores = q(64x64) @ k^T
  f8v zero = {};
  f8v sc[4] = {zero, zero, zero, zero};
  {
    const _Float16* qp = q_lds + (mt * 16 + col0) * KS64 + grp8;
    h16v a0 = frag_ld(qp), a1 = frag_ld(qp + 32);
#pragma unroll
    for (int nt = 0; nt < 4; ++nt) {
      const _Float16* kp = k_lds + (nt * 16 + col0) * KS64 + grp8;
      h16v k0 = frag_ld(kp), k1 = frag_ld(kp + 32);
      sc[nt] = wmma_f16(a0, k0, sc[nt]);
      sc[nt] = wmma_f16(a1, k1, sc[nt]);
    }
  }

  // Row softmax: row (i + 8*(lane>=16)) of strip lives in one 16-lane half.
#pragma unroll
  for (int i = 0; i < 8; ++i) {
    float m = fmaxf(fmaxf(sc[0][i], sc[1][i]), fmaxf(sc[2][i], sc[3][i]));
#pragma unroll
    for (int off = 8; off >= 1; off >>= 1)
      m = fmaxf(m, __shfl_xor(m, off, 32));
    float s = 0.0f;
#pragma unroll
    for (int nt = 0; nt < 4; ++nt) {
      float e = __expf(sc[nt][i] - m);
      sc[nt][i] = e;
      s += e;
    }
#pragma unroll
    for (int off = 8; off >= 1; off >>= 1)
      s += __shfl_xor(s, off, 32);
    float inv = __frcp_rn(s);
#pragma unroll
    for (int nt = 0; nt < 4; ++nt) sc[nt][i] *= inv;
  }

  // attnT[(b*NH+h)][z][s] = P[s][z]
  _Float16* dst = attnT + (size_t)(b * NH + h) * PANEL;
#pragma unroll
  for (int nt = 0; nt < 4; ++nt) {
    int z = nt * 16 + col0;
#pragma unroll
    for (int i = 0; i < 8; ++i)
      dst[z * 64 + mt * 16 + rb + i] = (_Float16)sc[nt][i];
  }
}

// ---------------------------------------------------------------------------
// Kernel 2: fused v-proj -> attn-apply -> out-proj -> +bias +residual.
// 256 blocks of 64 rows, 8 waves (wave = 16-row strip x 2 col tiles).
// ---------------------------------------------------------------------------
__global__ __launch_bounds__(256) void fused_out_kernel(
    const float* __restrict__ y, const _Float16* __restrict__ WvT,
    const float* __restrict__ bkv, const _Float16* __restrict__ WoT,
    const float* __restrict__ bo, const _Float16* __restrict__ attnT,
    float* __restrict__ out) {
  extern __shared__ char smem[];
  _Float16* y_lds = (_Float16*)smem;        // [64][776]
  _Float16* o_lds = y_lds + 64 * KSY;       // [64][776]
  _Float16* w_lds = o_lds + 64 * KSY;       // [64][72]
  _Float16* v_lds = w_lds + 64 * KS64;      // [64][72]
  _Float16* p_lds = v_lds + 64 * KS64;      // [64][72]

  const int tile = blockIdx.x;
  const int b = tile >> 5;
  const int n0 = (tile & 31) * 64;
  const int tid = threadIdx.x, lane = tid & 31, wave = tid >> 5;
  const int mt = wave & 3;
  const int nt0 = (wave >> 2) * 2, nt1 = nt0 + 1;
  const int col0 = lane & 15, grp8 = (lane >> 4) * 8, rb = (lane >> 4) * 8;

  const float* ytile = y + ((size_t)b * SEQ + n0) * DIM;

  // Stage y tile f32->f16 (once; reused by all 12 head chunks).
  for (int e = tid; e < 64 * DIM; e += 256) {
    int r = e / DIM, c = e - r * DIM;
    y_lds[r * KSY + c] = (_Float16)ytile[(size_t)r * DIM + c];
  }

  f8v zero = {};

  // ---------------- stages 1+2 per head chunk -----------------------------
  for (int jn = 0; jn < NH; ++jn) {
    f8v acc0 = zero, acc1 = zero;
    for (int ks = 0; ks < DIM / 64; ++ks) {
      __syncthreads();
      stage_panel(w_lds, WvT + (size_t)(jn * 12 + ks) * PANEL, tid, 256);
      if (ks + 1 < DIM / 64)
        __builtin_prefetch(WvT + (size_t)(jn * 12 + ks + 1) * PANEL + tid * 8,
                           0, 0);
      __syncthreads();
      const _Float16* yp = y_lds + (mt * 16 + col0) * KSY + ks * 64 + grp8;
      h16v a0 = frag_ld(yp), a1 = frag_ld(yp + 32);
      const _Float16* w0p = w_lds + (nt0 * 16 + col0) * KS64 + grp8;
      const _Float16* w1p = w_lds + (nt1 * 16 + col0) * KS64 + grp8;
      h16v b00 = frag_ld(w0p), b01 = frag_ld(w0p + 32);
      h16v b10 = frag_ld(w1p), b11 = frag_ld(w1p + 32);
      acc0 = wmma_f16(a0, b00, acc0);
      acc0 = wmma_f16(a1, b01, acc0);
      acc1 = wmma_f16(a0, b10, acc1);
      acc1 = wmma_f16(a1, b11, acc1);
    }
    __syncthreads();
    // v bias + store v_chunk (A layout [n_row][d]) as f16
    {
      float bv0 = bkv[DIM + jn * 64 + nt0 * 16 + col0];
      float bv1 = bkv[DIM + jn * 64 + nt1 * 16 + col0];
#pragma unroll
      for (int i = 0; i < 8; ++i) {
        v_lds[(mt * 16 + rb + i) * KS64 + nt0 * 16 + col0] =
            (_Float16)(acc0[i] + bv0);
        v_lds[(mt * 16 + rb + i) * KS64 + nt1 * 16 + col0] =
            (_Float16)(acc1[i] + bv1);
      }
    }
    // stage P^T tile (f16, panel-contiguous -> async copy)
    stage_panel(p_lds, attnT + (size_t)(b * NH + jn) * PANEL, tid, 256);
    __syncthreads();
    // o_chunk = v_chunk(64x64) @ P(64x64)
    {
      f8v oc0 = zero, oc1 = zero;
      const _Float16* vp = v_lds + (mt * 16 + col0) * KS64 + grp8;
      h16v a0 = frag_ld(vp), a1 = frag_ld(vp + 32);
      const _Float16* p0 = p_lds + (nt0 * 16 + col0) * KS64 + grp8;
      const _Float16* p1 = p_lds + (nt1 * 16 + col0) * KS64 + grp8;
      h16v b00 = frag_ld(p0), b01 = frag_ld(p0 + 32);
      h16v b10 = frag_ld(p1), b11 = frag_ld(p1 + 32);
      oc0 = wmma_f16(a0, b00, oc0);
      oc0 = wmma_f16(a1, b01, oc0);
      oc1 = wmma_f16(a0, b10, oc1);
      oc1 = wmma_f16(a1, b11, oc1);
#pragma unroll
      for (int i = 0; i < 8; ++i) {
        o_lds[(mt * 16 + rb + i) * KSY + jn * 64 + nt0 * 16 + col0] =
            (_Float16)oc0[i];
        o_lds[(mt * 16 + rb + i) * KSY + jn * 64 + nt1 * 16 + col0] =
            (_Float16)oc1[i];
      }
    }
  }
  __syncthreads();

  // ---------------- stage 3: final = o_tile @ Wo + bo + y ------------------
  float* otile = out + ((size_t)b * SEQ + n0) * DIM;
  for (int jo = 0; jo < NH; ++jo) {
    f8v acc0 = zero, acc1 = zero;
    for (int ks = 0; ks < DIM / 64; ++ks) {
      __syncthreads();
      stage_panel(w_lds, WoT + (size_t)(jo * 12 + ks) * PANEL, tid, 256);
      if (ks + 1 < DIM / 64)
        __builtin_prefetch(WoT + (size_t)(jo * 12 + ks + 1) * PANEL + tid * 8,
                           0, 0);
      __syncthreads();
      const _Float16* op = o_lds + (mt * 16 + col0) * KSY + ks * 64 + grp8;
      h16v a0 = frag_ld(op), a1 = frag_ld(op + 32);
      const _Float16* w0p = w_lds + (nt0 * 16 + col0) * KS64 + grp8;
      const _Float16* w1p = w_lds + (nt1 * 16 + col0) * KS64 + grp8;
      h16v b00 = frag_ld(w0p), b01 = frag_ld(w0p + 32);
      h16v b10 = frag_ld(w1p), b11 = frag_ld(w1p + 32);
      acc0 = wmma_f16(a0, b00, acc0);
      acc0 = wmma_f16(a1, b01, acc0);
      acc1 = wmma_f16(a0, b10, acc1);
      acc1 = wmma_f16(a1, b11, acc1);
    }
    {
      int c0 = jo * 64 + nt0 * 16 + col0;
      int c1 = jo * 64 + nt1 * 16 + col0;
      float bv0 = bo[c0], bv1 = bo[c1];
#pragma unroll
      for (int i = 0; i < 8; ++i) {
        int r = mt * 16 + rb + i;
        otile[(size_t)r * DIM + c0] =
            acc0[i] + bv0 + ytile[(size_t)r * DIM + c0];
        otile[(size_t)r * DIM + c1] =
            acc1[i] + bv1 + ytile[(size_t)r * DIM + c1];
      }
    }
  }
}

// ---------------------------------------------------------------------------
extern "C" void kernel_launch(void* const* d_in, const int* in_sizes, int n_in,
                              void* d_out, int out_size, void* d_ws,
                              size_t ws_size, hipStream_t stream) {
  const float* x   = (const float*)d_in[0];
  const float* y   = (const float*)d_in[1];
  const float* Wq  = (const float*)d_in[2];
  const float* bq  = (const float*)d_in[3];
  const float* Wkv = (const float*)d_in[4];
  const float* bkv = (const float*)d_in[5];
  const float* Wo  = (const float*)d_in[6];
  const float* bo  = (const float*)d_in[7];
  float* out = (float*)d_out;

  // workspace layout (halves): attnT | WqT | WkT | WvT | WoT  (~5.5 MB)
  _Float16* attnT = (_Float16*)d_ws;
  _Float16* WqT = attnT + (size_t)NB * NH * PANEL;
  _Float16* WkT = WqT + MATP;
  _Float16* WvT = WkT + MATP;
  _Float16* WoT = WvT + MATP;

  const size_t smem1 = (size_t)4 * 64 * KS64 * sizeof(_Float16);  // ~36 KB
  const size_t smem2 =
      ((size_t)2 * 64 * KSY + (size_t)3 * 64 * KS64) * sizeof(_Float16);

  (void)hipFuncSetAttribute((const void*)attn_prob_kernel,
                            hipFuncAttributeMaxDynamicSharedMemorySize,
                            (int)smem1);
  (void)hipFuncSetAttribute((const void*)fused_out_kernel,
                            hipFuncAttributeMaxDynamicSharedMemorySize,
                            (int)smem2);

  prep_weights<<<dim3((4 * MATP + 255) / 256), dim3(256), 0, stream>>>(
      Wq, Wkv, Wo, WqT, WkT, WvT, WoT);
  attn_prob_kernel<<<dim3(NB * NH), dim3(128), smem1, stream>>>(
      x, y, WqT, bq, WkT, bkv, attnT);
  fused_out_kernel<<<dim3(NB * (SEQ / 64)), dim3(256), smem2, stream>>>(
      y, WvT, bkv, WoT, bo, attnT, out);
}
